// Block_71339406787330
// MI455X (gfx1250) — matrix operand order repaired and checked by reading, not compiled, not run on hardware
//
#include <hip/hip_runtime.h>
#include <hip/hip_bf16.h>

typedef __bf16 bf16;
typedef __bf16 v16bf __attribute__((ext_vector_type(16)));
typedef float  v8f   __attribute__((ext_vector_type(8)));

union AF {
  v16bf v;
  bf16  h[16];
  uint4 q[2];
};

__device__ inline v8f wmma_bf16f32(const AF& a, const AF& b, v8f c) {
#if defined(__HIP_DEVICE_COMPILE__)
  // D = A(16x32 bf16) * B(32x16 bf16) + C(16x16 f32)
  return __builtin_amdgcn_wmma_f32_16x16x32_bf16(false, a.v, false, b.v, (short)0, c,
                                                 false, false);
#else
  return c;
#endif
}

// CDNA5 async global->LDS copy (ASYNCcnt tracked, no VGPR round-trip).
// lds_off: byte offset within workgroup LDS (low 32 bits of generic ptr).
__device__ inline void async_b128(unsigned int lds_off, const void* gaddr) {
#if defined(__HIP_DEVICE_COMPILE__)
  asm volatile("global_load_async_to_lds_b128 %0, %1, off"
               :: "v"(lds_off), "v"(gaddr)
               : "memory");
#endif
}
__device__ inline void wait_async_10() {
#if defined(__HIP_DEVICE_COMPILE__)
  asm volatile("s_wait_asynccnt 0xa" ::: "memory");
#endif
}
__device__ inline void wait_async_0() {
#if defined(__HIP_DEVICE_COMPILE__)
  asm volatile("s_wait_asynccnt 0x0" ::: "memory");
#endif
}

__device__ inline unsigned int lds_off_u32(const void* p) {
  return (unsigned int)(uintptr_t)p;  // LDS aperture: addr[31:0] is the LDS offset
}

// ---------------------------------------------------------------------------
// Weight prep: f32 -> bf16, layout preserved (W[OUT][IN]; rows are K-contiguous,
// which is exactly the WMMA B-fragment layout -> no transpose needed anywhere).
// ---------------------------------------------------------------------------
__launch_bounds__(256)
__global__ void cvt_bf16_kernel(const float4* __restrict__ W, bf16* __restrict__ Wb,
                                int total4) {
  int idx = blockIdx.x * 256 + threadIdx.x;
  if (idx >= total4) return;
  float4 f = W[idx];
  union { bf16 h[4]; uint2 u; } o;
  o.h[0] = (bf16)f.x; o.h[1] = (bf16)f.y; o.h[2] = (bf16)f.z; o.h[3] = (bf16)f.w;
  *(uint2*)&Wb[(size_t)idx * 4] = o.u;
}

// ---------------------------------------------------------------------------
// LayerNorm over 768, writes bf16; optional remap from natural (b,d,h,w) row
// order to window-partitioned order (win*512 + tok).
// ---------------------------------------------------------------------------
__launch_bounds__(256)
__global__ void ln_kernel(const float* __restrict__ x, const float* __restrict__ w,
                          const float* __restrict__ b, bf16* __restrict__ out,
                          int to_window) {
  __shared__ float r1[256];
  __shared__ float r2[256];
  int row = blockIdx.x;
  int tid = threadIdx.x;
  const float* xr = x + (size_t)row * 768;
  float v0 = xr[tid], v1 = xr[tid + 256], v2 = xr[tid + 512];
  r1[tid] = v0 + v1 + v2;
  r2[tid] = v0 * v0 + v1 * v1 + v2 * v2;
  __syncthreads();
  for (int off = 128; off > 0; off >>= 1) {
    if (tid < off) { r1[tid] += r1[tid + off]; r2[tid] += r2[tid + off]; }
    __syncthreads();
  }
  float mean = r1[0] * (1.0f / 768.0f);
  float var  = r2[0] * (1.0f / 768.0f) - mean * mean;
  float inv  = rsqrtf(var + 1e-5f);
  size_t orow = (size_t)row;
  if (to_window) {
    int b0 = row >> 12;
    int d  = (row >> 8) & 15;
    int h  = (row >> 4) & 15;
    int ww = row & 15;
    int win = ((b0 * 2 + (d >> 3)) * 2 + (h >> 3)) * 2 + (ww >> 3);
    int tok = ((d & 7) << 6) | ((h & 7) << 3) | (ww & 7);
    orow = (size_t)win * 512 + tok;
  }
  bf16* orp = out + orow * 768;
  orp[tid]       = (bf16)((v0 - mean) * inv * w[tid]       + b[tid]);
  orp[tid + 256] = (bf16)((v1 - mean) * inv * w[tid + 256] + b[tid + 256]);
  orp[tid + 512] = (bf16)((v2 - mean) * inv * w[tid + 512] + b[tid + 512]);
}

// ---------------------------------------------------------------------------
// bf16 WMMA GEMM: C[M,N] = A[M,K] @ W[N,K]^T (+bias).
// Block tile 64x256, BK=64, 8 waves each own a 32x64 subtile (16 WMMA/tile).
// Double-buffered LDS tiles filled with GLOBAL_LOAD_ASYNC_TO_LDS_B128;
// pipeline fenced with s_wait_asynccnt (in-order per-wave completion).
// MODE 0: scatter QKV into per-head q/k/v buffers (p0/p1/p2)
// MODE 1: proj + window-unpartition + residual with pf0 (x), write pf1 (x2 f32)
// MODE 2: fc1 + exact GELU -> p0 bf16
// MODE 3: fc2 + bias + residual pf0 -> pf1 f32
// ---------------------------------------------------------------------------
#define TILE_BYTES (8192 + 32768)  // Al 64x64 bf16 + Bl 256x64 bf16

template <int MODE>
__launch_bounds__(256)
__global__ void gemm_bf16_kernel(const bf16* __restrict__ A, const bf16* __restrict__ B,
                                 const float* __restrict__ bias,
                                 int M, int N, int K,
                                 bf16* __restrict__ p0, bf16* __restrict__ p1,
                                 bf16* __restrict__ p2,
                                 const float* __restrict__ pf0, float* __restrict__ pf1) {
  (void)M;
  extern __shared__ char gsm[];  // 2 * TILE_BYTES

  const int tid  = threadIdx.x;
  const int wave = tid >> 5;
  const int lane = tid & 31;
  const int lm   = lane & 15;
  const int khf  = lane >> 4;
  const int wm   = wave >> 2;  // 0..1
  const int wn   = wave & 3;   // 0..3
  const int bm   = blockIdx.y * 64;
  const int bn   = blockIdx.x * 256;

  // issue 10 async b128 copies: A tile 64x64 (2/thread) + B tile 256x64 (8/thread)
  auto issue_tile = [&](int buf, int kk) {
    unsigned int al = lds_off_u32(gsm + buf * TILE_BYTES);
    unsigned int bl = al + 8192;
#pragma unroll
    for (int it = 0; it < 2; ++it) {
      int chunk = tid + it * 256;          // 0..511
      int r = chunk >> 3;                  // 0..63
      int c = (chunk & 7) * 8;             // 0..56
      async_b128(al + (unsigned)(r * 64 + c) * 2, &A[(size_t)(bm + r) * K + kk + c]);
    }
#pragma unroll
    for (int it = 0; it < 8; ++it) {
      int chunk = tid + it * 256;          // 0..2047
      int n = chunk >> 3;                  // 0..255
      int c = (chunk & 7) * 8;
      async_b128(bl + (unsigned)(n * 64 + c) * 2, &B[(size_t)(bn + n) * K + kk + c]);
    }
  };

  v8f acc[2][4];
  for (int i = 0; i < 2; ++i)
    for (int j = 0; j < 4; ++j)
      for (int r = 0; r < 8; ++r) acc[i][j][r] = 0.0f;

  issue_tile(0, 0);
  const int T = K >> 6;
  for (int t = 0; t < T; ++t) {
    if (t + 1 < T) {
      issue_tile((t + 1) & 1, (t + 1) << 6);
      wait_async_10();   // own asyncs in order: tile t complete, t+1 may fly
    } else {
      wait_async_0();
    }
    __syncthreads();     // every wave waited for its own copies -> tiles ready

    const bf16* Alp = (const bf16*)(gsm + (t & 1) * TILE_BYTES);
    const bf16* Blp = Alp + 4096;  // 8192 bytes
#pragma unroll
    for (int ks = 0; ks < 64; ks += 32) {
      AF fa[2], fb[4];
#pragma unroll
      for (int i = 0; i < 2; ++i) {
        const bf16* p = Alp + (wm * 32 + i * 16 + lm) * 64 + ks + khf * 8;
        fa[i].q[0] = *(const uint4*)p;
        fa[i].q[1] = *(const uint4*)(p + 16);
      }
#pragma unroll
      for (int j = 0; j < 4; ++j) {
        const bf16* p = Blp + (wn * 64 + j * 16 + lm) * 64 + ks + khf * 8;
        fb[j].q[0] = *(const uint4*)p;
        fb[j].q[1] = *(const uint4*)(p + 16);
      }
#pragma unroll
      for (int i = 0; i < 2; ++i)
#pragma unroll
        for (int j = 0; j < 4; ++j)
          acc[i][j] = wmma_bf16f32(fa[i], fb[j], acc[i][j]);
    }
    __syncthreads();     // all reads of this buffer done before it is refilled
  }

#pragma unroll
  for (int i = 0; i < 2; ++i)
#pragma unroll
    for (int j = 0; j < 4; ++j)
#pragma unroll
      for (int r = 0; r < 8; ++r) {
        int row = bm + wm * 32 + i * 16 + khf * 8 + r;
        int col = bn + wn * 64 + j * 16 + lm;
        float vacc = acc[i][j][r] + bias[col];
        if (MODE == 0) {
          int win = row >> 9, tok = row & 511;
          int sel = col / 768;
          int w2  = col - sel * 768;
          int head = w2 >> 6, hd = w2 & 63;
          size_t idx = (((size_t)(win * 12 + head)) * 512 + tok) * 64 + hd;
          bf16 val = (bf16)vacc;
          if (sel == 0)      p0[idx] = val;
          else if (sel == 1) p1[idx] = val;
          else               p2[idx] = val;
        } else if (MODE == 1) {
          int win = row >> 9, tok = row & 511;
          int b0 = win >> 3, dblk = (win >> 2) & 1, hblk = (win >> 1) & 1, wblk = win & 1;
          int dz = tok >> 6, hy = (tok >> 3) & 7, wx = tok & 7;
          int dd = dblk * 8 + dz, hh = hblk * 8 + hy, ww = wblk * 8 + wx;
          size_t nat = ((((size_t)b0 * 16 + dd) * 16 + hh) * 16 + ww) * 768 + col;
          pf1[nat] = vacc + pf0[nat];
        } else if (MODE == 2) {
          float g = 0.5f * vacc * (1.0f + erff(vacc * 0.70710678118654752f));
          p0[(size_t)row * N + col] = (bf16)g;
        } else {
          size_t idx = (size_t)row * N + col;
          pf1[idx] = vacc + pf0[idx];
        }
      }
}

// ---------------------------------------------------------------------------
// Fused attention: per (window*head, 32-row block):
//   S = scale*(q @ k^T) + rel-pos bias ; softmax rows ; out = P @ v
// q,k,v are bf16 [win][head][tok][64]. attnout bf16 [8192][768].
// ---------------------------------------------------------------------------
__launch_bounds__(256)
__global__ void attn_kernel(const bf16* __restrict__ qg, const bf16* __restrict__ kg,
                            const bf16* __restrict__ vg,
                            const float* __restrict__ rpd, const float* __restrict__ rph,
                            const float* __restrict__ rpw,
                            bf16* __restrict__ attnout) {
  extern __shared__ char smem[];
  float* S    = (float*)smem;                              // 32*512 f32   (64KB)
  bf16*  vT   = (bf16*)(smem + 32 * 512 * 4);              // 64*512 bf16  (64KB)
  float* reld = (float*)(smem + 32 * 512 * 4 + 64 * 512 * 2);  // 32*8
  float* relh = reld + 32 * 8;
  float* relw = relh + 32 * 8;
  float* rinv = relw + 32 * 8;   // 32
  float* outl = rinv + 32;       // 32*64 f32

  const int tid  = threadIdx.x;
  const int wave = tid >> 5;
  const int lane = tid & 31;
  const int lm   = lane & 15;
  const int khf  = lane >> 4;

  const int wh  = blockIdx.x;  // win*12 + head
  const int rb  = blockIdx.y;  // 32-row block, 0..15
  const int win = wh / 12, head = wh % 12;

  const bf16* qh = qg + (size_t)wh * 512 * 64;
  const bf16* kh = kg + (size_t)wh * 512 * 64;
  const bf16* vh = vg + (size_t)wh * 512 * 64;

  for (int idx = tid; idx < 32 * 64; idx += 256) outl[idx] = 0.0f;

  // stage V transposed: vT[hd][tok]
#pragma unroll
  for (int it = 0; it < 16; ++it) {
    int flat = tid + it * 256;
    int tok = flat >> 3;
    int hc  = (flat & 7) * 8;
    uint4 dv = *(const uint4*)&vh[tok * 64 + hc];
    const bf16* pv = (const bf16*)&dv;
#pragma unroll
    for (int e = 0; e < 8; ++e) vT[(hc + e) * 512 + tok] = pv[e];
  }

  // decomposed rel-pos bias tables: reld/relh/relw[row][k-coord]
  {
    int r = tid >> 3, j = tid & 7;
    int tok = rb * 32 + r;
    int qd = tok >> 6, qy = (tok >> 3) & 7, qx = tok & 7;
    const bf16* qrow = qh + tok * 64;
    const float* pd = rpd + (qd - j + 7) * 64;
    const float* ph = rph + (qy - j + 7) * 64;
    const float* pw = rpw + (qx - j + 7) * 64;
    float sd = 0.f, sh = 0.f, sw = 0.f;
    for (int c = 0; c < 64; ++c) {
      float qc = (float)qrow[c];
      sd += qc * pd[c]; sh += qc * ph[c]; sw += qc * pw[c];
    }
    reld[r * 8 + j] = sd; relh[r * 8 + j] = sh; relw[r * 8 + j] = sw;
  }

  // scores: rows rb*32..+31, this wave covers cols n0..n0+63, K=64
  v8f acc[2][4];
  for (int i = 0; i < 2; ++i)
    for (int j = 0; j < 4; ++j)
      for (int r = 0; r < 8; ++r) acc[i][j][r] = 0.0f;

  const int n0 = wave * 64;
#pragma unroll
  for (int ks = 0; ks < 2; ++ks) {
    int k0 = ks * 32;
    AF fa[2], fb[4];
#pragma unroll
    for (int i = 0; i < 2; ++i) {
      const bf16* p = qh + (size_t)(rb * 32 + i * 16 + lm) * 64 + k0 + khf * 8;
      fa[i].q[0] = *(const uint4*)p;
      fa[i].q[1] = *(const uint4*)(p + 16);
    }
#pragma unroll
    for (int j = 0; j < 4; ++j) {
      const bf16* p = kh + (size_t)(n0 + j * 16 + lm) * 64 + k0 + khf * 8;
      fb[j].q[0] = *(const uint4*)p;
      fb[j].q[1] = *(const uint4*)(p + 16);
    }
#pragma unroll
    for (int i = 0; i < 2; ++i)
#pragma unroll
      for (int j = 0; j < 4; ++j)
        acc[i][j] = wmma_bf16f32(fa[i], fb[j], acc[i][j]);
  }

  __syncthreads();  // vT + rel tables + outl zero ready

  // scale + bias -> S
#pragma unroll
  for (int i = 0; i < 2; ++i)
#pragma unroll
    for (int j = 0; j < 4; ++j)
#pragma unroll
      for (int r = 0; r < 8; ++r) {
        int ml = i * 16 + khf * 8 + r;
        int n  = n0 + j * 16 + lm;
        int kd = n >> 6, ky = (n >> 3) & 7, kx = n & 7;
        S[ml * 512 + n] = acc[i][j][r] * 0.125f
                        + reld[ml * 8 + kd] + relh[ml * 8 + ky] + relw[ml * 8 + kx];
      }
  __syncthreads();

  // softmax: each wave owns 4 rows
  for (int rr = 0; rr < 4; ++rr) {
    int r = wave * 4 + rr;
    float mx = -1e30f;
    for (int c = lane; c < 512; c += 32) mx = fmaxf(mx, S[r * 512 + c]);
#pragma unroll
    for (int off = 16; off > 0; off >>= 1) mx = fmaxf(mx, __shfl_xor(mx, off, 32));
    float sum = 0.0f;
    for (int c = lane; c < 512; c += 32) {
      float e = __expf(S[r * 512 + c] - mx);
      S[r * 512 + c] = e;
      sum += e;
    }
#pragma unroll
    for (int off = 16; off > 0; off >>= 1) sum += __shfl_xor(sum, off, 32);
    if (lane == 0) rinv[r] = 1.0f / sum;
  }
  __syncthreads();

  // P @ V: wave handles K slice [wave*64, wave*64+64); reduce via LDS atomics
  v8f acc2[2][4];
  for (int i = 0; i < 2; ++i)
    for (int j = 0; j < 4; ++j)
      for (int r = 0; r < 8; ++r) acc2[i][j][r] = 0.0f;

#pragma unroll
  for (int ks = 0; ks < 2; ++ks) {
    int kbase = wave * 64 + ks * 32;
    AF fa2[2], fb2[4];
#pragma unroll
    for (int i = 0; i < 2; ++i) {
      int m = i * 16 + lm;
      float inv = rinv[m];
      const float* sp = S + m * 512 + kbase + khf * 8;
#pragma unroll
      for (int e = 0; e < 8; ++e) fa2[i].h[e]     = (bf16)(sp[e] * inv);
#pragma unroll
      for (int e = 0; e < 8; ++e) fa2[i].h[8 + e] = (bf16)(sp[16 + e] * inv);
    }
#pragma unroll
    for (int j = 0; j < 4; ++j) {
      const bf16* p = vT + (j * 16 + lm) * 512 + kbase + khf * 8;
      fb2[j].q[0] = *(const uint4*)p;
      fb2[j].q[1] = *(const uint4*)(p + 16);
    }
#pragma unroll
    for (int i = 0; i < 2; ++i)
#pragma unroll
      for (int j = 0; j < 4; ++j)
        acc2[i][j] = wmma_bf16f32(fa2[i], fb2[j], acc2[i][j]);
  }

#pragma unroll
  for (int i = 0; i < 2; ++i)
#pragma unroll
    for (int j = 0; j < 4; ++j)
#pragma unroll
      for (int r = 0; r < 8; ++r)
        atomicAdd(&outl[(i * 16 + khf * 8 + r) * 64 + j * 16 + lm], acc2[i][j][r]);
  __syncthreads();

  for (int idx = tid; idx < 32 * 64; idx += 256) {
    int ml = idx >> 6, n = idx & 63;
    int tokg = rb * 32 + ml;
    attnout[((size_t)(win * 512 + tokg)) * 768 + head * 64 + n] = (bf16)outl[idx];
  }
}

// ---------------------------------------------------------------------------
extern "C" void kernel_launch(void* const* d_in, const int* in_sizes, int n_in,
                              void* d_out, int out_size, void* d_ws, size_t ws_size,
                              hipStream_t stream) {
  (void)in_sizes; (void)n_in; (void)out_size; (void)ws_size;
  const float* x      = (const float*)d_in[0];
  const float* ln1_w  = (const float*)d_in[1];
  const float* ln1_b  = (const float*)d_in[2];
  const float* qkv_w  = (const float*)d_in[3];
  const float* qkv_b  = (const float*)d_in[4];
  const float* proj_w = (const float*)d_in[5];
  const float* proj_b = (const float*)d_in[6];
  const float* rpd    = (const float*)d_in[7];
  const float* rph    = (const float*)d_in[8];
  const float* rpw    = (const float*)d_in[9];
  const float* ln2_w  = (const float*)d_in[10];
  const float* ln2_b  = (const float*)d_in[11];
  const float* fc1_w  = (const float*)d_in[12];
  const float* fc1_b  = (const float*)d_in[13];
  const float* fc2_w  = (const float*)d_in[14];
  const float* fc2_b  = (const float*)d_in[15];

  char* ws = (char*)d_ws;
  size_t off = 0;
  auto carve = [&](size_t bytes) -> void* {
    void* p = ws + off;
    off += (bytes + 255) & ~(size_t)255;
    return p;
  };
  bf16*  wb_qkv  = (bf16*)carve((size_t)2304 * 768 * 2);   // [N][K] bf16, no transpose
  bf16*  wb_proj = (bf16*)carve((size_t)768 * 768 * 2);
  bf16*  wb_fc1  = (bf16*)carve((size_t)3072 * 768 * 2);
  bf16*  wb_fc2  = (bf16*)carve((size_t)768 * 3072 * 2);
  bf16*  xw      = (bf16*)carve((size_t)8192 * 768 * 2);
  bf16*  qb      = (bf16*)carve((size_t)16 * 12 * 512 * 64 * 2);
  bf16*  kb      = (bf16*)carve((size_t)16 * 12 * 512 * 64 * 2);
  bf16*  vb      = (bf16*)carve((size_t)16 * 12 * 512 * 64 * 2);
  bf16*  attno   = (bf16*)carve((size_t)8192 * 768 * 2);
  float* x2      = (float*)carve((size_t)8192 * 768 * 4);
  bf16*  m2      = (bf16*)carve((size_t)8192 * 768 * 2);
  bf16*  h1      = (bf16*)carve((size_t)8192 * 3072 * 2);

  dim3 blk(256);

  cvt_bf16_kernel<<<(2304 * 768 / 4 + 255) / 256, blk, 0, stream>>>(
      (const float4*)qkv_w, wb_qkv, 2304 * 768 / 4);
  cvt_bf16_kernel<<<(768 * 768 / 4 + 255) / 256, blk, 0, stream>>>(
      (const float4*)proj_w, wb_proj, 768 * 768 / 4);
  cvt_bf16_kernel<<<(3072 * 768 / 4 + 255) / 256, blk, 0, stream>>>(
      (const float4*)fc1_w, wb_fc1, 3072 * 768 / 4);
  cvt_bf16_kernel<<<(768 * 3072 / 4 + 255) / 256, blk, 0, stream>>>(
      (const float4*)fc2_w, wb_fc2, 768 * 3072 / 4);

  // LN1 + window partition -> xw (bf16)
  ln_kernel<<<8192, blk, 0, stream>>>(x, ln1_w, ln1_b, xw, 1);

  constexpr int GEMM_SMEM = 2 * TILE_BYTES;  // 80KB double-buffered
  hipFuncSetAttribute(reinterpret_cast<const void*>(gemm_bf16_kernel<0>),
                      hipFuncAttributeMaxDynamicSharedMemorySize, GEMM_SMEM);
  hipFuncSetAttribute(reinterpret_cast<const void*>(gemm_bf16_kernel<1>),
                      hipFuncAttributeMaxDynamicSharedMemorySize, GEMM_SMEM);
  hipFuncSetAttribute(reinterpret_cast<const void*>(gemm_bf16_kernel<2>),
                      hipFuncAttributeMaxDynamicSharedMemorySize, GEMM_SMEM);
  hipFuncSetAttribute(reinterpret_cast<const void*>(gemm_bf16_kernel<3>),
                      hipFuncAttributeMaxDynamicSharedMemorySize, GEMM_SMEM);

  // QKV: [8192,768] @ W[2304,768]^T -> q/k/v per-head bf16
  gemm_bf16_kernel<0><<<dim3(9, 128), blk, GEMM_SMEM, stream>>>(
      xw, wb_qkv, qkv_b, 8192, 2304, 768, qb, kb, vb, nullptr, nullptr);

  // fused windowed attention
  constexpr int ASMEM = 32 * 512 * 4 + 64 * 512 * 2 + 3 * 32 * 8 * 4 + 32 * 4 + 32 * 64 * 4;
  hipFuncSetAttribute(reinterpret_cast<const void*>(attn_kernel),
                      hipFuncAttributeMaxDynamicSharedMemorySize, ASMEM);
  attn_kernel<<<dim3(192, 16), blk, ASMEM, stream>>>(qb, kb, vb, rpd, rph, rpw, attno);

  // proj + unpartition + residual -> x2 (f32)
  gemm_bf16_kernel<1><<<dim3(3, 128), blk, GEMM_SMEM, stream>>>(
      attno, wb_proj, proj_b, 8192, 768, 768, nullptr, nullptr, nullptr, x, x2);

  // LN2 -> m2 (bf16)
  ln_kernel<<<8192, blk, 0, stream>>>(x2, ln2_w, ln2_b, m2, 0);

  // FC1 + GELU -> h1 (bf16)
  gemm_bf16_kernel<2><<<dim3(12, 128), blk, GEMM_SMEM, stream>>>(
      m2, wb_fc1, fc1_b, 8192, 3072, 768, h1, nullptr, nullptr, nullptr, nullptr);

  // FC2 + bias + residual -> out (f32)
  gemm_bf16_kernel<3><<<dim3(3, 128), blk, GEMM_SMEM, stream>>>(
      h1, wb_fc2, fc2_b, 8192, 768, 3072, nullptr, nullptr, nullptr, x2, (float*)d_out);
}
